// Architecture_4346506903935
// MI455X (gfx1250) — matrix operand order repaired and checked by reading, not compiled, not run on hardware
//
#include <hip/hip_runtime.h>
#include <hip/hip_bf16.h>
#include <math.h>

// ---- problem constants -----------------------------------------------------
constexpr int B_  = 32;
constexpr int S_  = 1024;
constexpr int D_  = 512;
constexpr int H_  = 8;
constexpr int L_  = 4;
constexpr int DFF_ = 2048;
constexpr int DK_ = 64;               // D / H
constexpr int MS_ = B_ * S_;          // 32768 rows

typedef __bf16 bf16;
typedef __attribute__((ext_vector_type(16))) bf16  v16bf;
typedef __attribute__((ext_vector_type(8)))  float v8f;
typedef __attribute__((ext_vector_type(4)))  unsigned int u32x4;
typedef __attribute__((ext_vector_type(8)))  int i32x8;
typedef __attribute__((ext_vector_type(4)))  int i32x4;

#define DEV __device__ __forceinline__

// ---- TDM availability probing ----------------------------------------------
#if defined(__has_builtin)
#if __has_builtin(__builtin_amdgcn_tensor_load_to_lds) && \
    __has_builtin(__builtin_amdgcn_s_wait_tensorcnt)
#define USE_TDM 1
#endif
#endif

DEV v8f wmma_bf16(v16bf a, v16bf b, v8f c) {
    return __builtin_amdgcn_wmma_f32_16x16x32_bf16(
        false, a, false, b, (short)0, c, false, false);
}

// ---- WMMA fragment loaders (contiguous k = 16h + s order, same in A and B) -----------------
// A matrix 16x32 (MxK), source row-major: elem(m,k) = base[m*ld + k]
DEV v16bf load_a_frag(const bf16* base, int ld) {
    int lane = threadIdx.x & 31;
    int m = lane & 15, half = lane >> 4;       // lanes 0-15: K 0-15; 16-31: K 16-31
    const bf16* rp = base + m * ld + half * 16;
    v16bf a;
#pragma unroll
    for (int r = 0; r < 4; ++r) { a[2*r] = rp[2*r]; a[2*r+1] = rp[2*r+1]; }
    rp += 8;
#pragma unroll
    for (int r = 0; r < 4; ++r) { a[8+2*r] = rp[2*r]; a[8+2*r+1] = rp[2*r+1]; }
    return a;
}

// B matrix 32x16 (KxN), source N-major (e.g. W[N][K]): elem(k,n) = base[n*ld + k]
DEV v16bf load_b_frag(const bf16* base, int ld) {
    int lane = threadIdx.x & 31;
    int n = lane & 15, kb = (lane >> 4) * 16;  // lanes 0-15: K 0-15; 16-31: K 16-31
    const bf16* rp = base + n * ld + kb;
    v16bf b;
#pragma unroll
    for (int r = 0; r < 8; ++r) { b[2*r] = rp[2*r]; b[2*r+1] = rp[2*r+1]; }
    return b;
}

// B matrix 32x16, source K-major (e.g. V[K][N] rows): elem(k,n) = base[k*ldk + n]
DEV v16bf load_b_frag_km(const bf16* base, int ldk) {
    int lane = threadIdx.x & 31;
    int n = lane & 15, kb = (lane >> 4) * 16;
    v16bf b;
#pragma unroll
    for (int r = 0; r < 8; ++r) {
        b[2*r]   = base[(kb + 2*r    ) * ldk + n];
        b[2*r+1] = base[(kb + 2*r + 1) * ldk + n];
    }
    return b;
}

DEV float rowmax16(float v) {
#pragma unroll
    for (int m = 1; m <= 8; m <<= 1) v = fmaxf(v, __shfl_xor(v, m, 32));
    return v;
}
DEV float rowsum16(float v) {
#pragma unroll
    for (int m = 1; m <= 8; m <<= 1) v += __shfl_xor(v, m, 32);
    return v;
}

#ifdef USE_TDM
// ---- Tensor Data Mover: 2D tile load, bf16 (data_size=2B) ------------------
// Moves tile_w x tile_h (K-contig x rows) from global (row stride =
// stride_elems) into LDS at lds_off. pad=true inserts 4 DWORDs of LDS padding
// per 16 DWORDs stored -> LDS row stride = 40 bf16 for a 32-elem tile row.
DEV void tdm_load_2d(const void* gptr, unsigned lds_off,
                     unsigned tensor_w, unsigned tensor_h,
                     unsigned tile_w, unsigned tile_h,
                     unsigned stride_elems, bool pad) {
    unsigned long long ga = (unsigned long long)gptr;
    u32x4 g0;
    g0[0] = 1u;                                             // count=1, user mode
    g0[1] = lds_off;                                        // lds_addr (bytes)
    g0[2] = (unsigned)(ga & 0xffffffffu);                   // global_addr[31:0]
    g0[3] = (unsigned)((ga >> 32) & 0x01ffffffu) | (2u << 30); // addr[56:32] | type=2
    i32x8 g1;
    unsigned flags = (1u << 16);                            // data_size = 2 bytes
    if (pad) flags |= (1u << 20) | (3u << 22) | (3u << 25); // pad: every 16 DW, +4 DW
    g1[0] = (int)flags;
    g1[1] = (int)((tensor_w & 0xffffu) << 16);              // tensor_dim0[15:0]
    g1[2] = (int)((tensor_w >> 16) | ((tensor_h & 0xffffu) << 16));
    g1[3] = (int)((tensor_h >> 16) | (tile_w << 16));       // tile_dim0
    g1[4] = (int)tile_h;                                    // tile_dim1 (tile_dim2=0)
    g1[5] = (int)stride_elems;                              // tensor_dim0_stride[31:0]
    g1[6] = 0;
    g1[7] = 0;
    i32x4 z = {};
#if __has_include(<hip/amd_detail/amd_gfx1250_TDM.h>)
    i32x8 z8 = {};
    __builtin_amdgcn_tensor_load_to_lds(g0, g1, z, z, z8, 0);   // clang-23 form
#else
    __builtin_amdgcn_tensor_load_to_lds(g0, g1, z, z, 0);       // ROCm 7.2 form
#endif
}
DEV unsigned lds_offset_of(const void* p) {                 // generic ptr -> LDS byte offset
    return (unsigned)(unsigned long long)p;
}
#endif

// ---- prep: x = q+pe, y = qa+pe; fp32 + bf16 copies -------------------------
__global__ __launch_bounds__(256) void prep_kernel(
    const float* __restrict__ q, const float* __restrict__ qa,
    const float* __restrict__ pe, float* __restrict__ xf,
    bf16* __restrict__ xb, bf16* __restrict__ yb) {
    size_t idx = (size_t)blockIdx.x * 256 + threadIdx.x;
    size_t p = idx % (size_t)(S_ * D_);
    float xv = q[idx]  + pe[p];
    float yv = qa[idx] + pe[p];
    xf[idx] = xv;
    xb[idx] = (bf16)xv;
    yb[idx] = (bf16)yv;
}

__global__ __launch_bounds__(256) void cast_kernel(
    const float* __restrict__ in, bf16* __restrict__ out) {
    size_t idx = (size_t)blockIdx.x * 256 + threadIdx.x;
    out[idx] = (bf16)in[idx];
}

// ---- WMMA GEMM: C[M,N] = A[M,K] @ W[N,K]^T + bias --------------------------
// block tile 128x128, 8 waves, each wave 32x64, K-step 32.
// LDS tiles double-buffered and filled by the Tensor Data Mover (wave 0 issues
// next-tile TENSOR_LOAD_TO_LDS while all waves run WMMAs on the current tile).
template<bool RELU, bool ATTN_LAYOUT>
__global__ __launch_bounds__(256) void gemm_wmma(
    const bf16* __restrict__ A, const bf16* __restrict__ W,
    const float* __restrict__ bias, float* __restrict__ Cf,
    bf16* __restrict__ Cb, int M, int N, int K) {
    constexpr int LDT = 40;                          // padded LDS stride (keeps b128 aligned)
    __shared__ bf16 As[2][128 * LDT];
    __shared__ bf16 Bs[2][128 * LDT];
    const int m0 = blockIdx.y * 128, n0 = blockIdx.x * 128;
    const int tid = threadIdx.x;
    const int wave = tid >> 5;
    const int wm = (wave >> 1) * 32;                 // 0,32,64,96
    const int wn = (wave & 1) * 64;                  // 0,64
    const bool issuer = (wave == 0);

    v8f acc[2][4];
#pragma unroll
    for (int i = 0; i < 2; ++i)
#pragma unroll
        for (int j = 0; j < 4; ++j) acc[i][j] = (v8f){};

#ifdef USE_TDM
    if (issuer) {
        tdm_load_2d(A + (size_t)m0 * K, lds_offset_of(&As[0][0]),
                    (unsigned)K, (unsigned)M, 32u, 128u, (unsigned)K, true);
        tdm_load_2d(W + (size_t)n0 * K, lds_offset_of(&Bs[0][0]),
                    (unsigned)K, (unsigned)N, 32u, 128u, (unsigned)K, true);
    }
#else
    const int lrow = tid >> 2;                       // 0..63
    const int lcol = (tid & 3) * 8;                  // 0,8,16,24
#endif

    int buf = 0;
    for (int kk = 0; kk < K; kk += 32) {
#ifdef USE_TDM
        if (issuer) __builtin_amdgcn_s_wait_tensorcnt(0);
        __syncthreads();
        if (issuer && (kk + 32 < K)) {               // prefetch next tile via TDM
            tdm_load_2d(A + (size_t)m0 * K + kk + 32, lds_offset_of(&As[buf ^ 1][0]),
                        (unsigned)K, (unsigned)M, 32u, 128u, (unsigned)K, true);
            tdm_load_2d(W + (size_t)n0 * K + kk + 32, lds_offset_of(&Bs[buf ^ 1][0]),
                        (unsigned)K, (unsigned)N, 32u, 128u, (unsigned)K, true);
        }
#else
        *(uint4*)(As[buf] + lrow      * LDT + lcol) = *(const uint4*)(A + (size_t)(m0 + lrow     ) * K + kk + lcol);
        *(uint4*)(As[buf] + (lrow+64) * LDT + lcol) = *(const uint4*)(A + (size_t)(m0 + lrow + 64) * K + kk + lcol);
        *(uint4*)(Bs[buf] + lrow      * LDT + lcol) = *(const uint4*)(W + (size_t)(n0 + lrow     ) * K + kk + lcol);
        *(uint4*)(Bs[buf] + (lrow+64) * LDT + lcol) = *(const uint4*)(W + (size_t)(n0 + lrow + 64) * K + kk + lcol);
        __syncthreads();
#endif
        v16bf a0 = load_a_frag(As[buf] + (wm +  0) * LDT, LDT);
        v16bf a1 = load_a_frag(As[buf] + (wm + 16) * LDT, LDT);
        v16bf b0 = load_b_frag(Bs[buf] + (wn +  0) * LDT, LDT);
        v16bf b1 = load_b_frag(Bs[buf] + (wn + 16) * LDT, LDT);
        v16bf b2 = load_b_frag(Bs[buf] + (wn + 32) * LDT, LDT);
        v16bf b3 = load_b_frag(Bs[buf] + (wn + 48) * LDT, LDT);
        acc[0][0] = wmma_bf16(a0, b0, acc[0][0]);
        acc[0][1] = wmma_bf16(a0, b1, acc[0][1]);
        acc[0][2] = wmma_bf16(a0, b2, acc[0][2]);
        acc[0][3] = wmma_bf16(a0, b3, acc[0][3]);
        acc[1][0] = wmma_bf16(a1, b0, acc[1][0]);
        acc[1][1] = wmma_bf16(a1, b1, acc[1][1]);
        acc[1][2] = wmma_bf16(a1, b2, acc[1][2]);
        acc[1][3] = wmma_bf16(a1, b3, acc[1][3]);
        __syncthreads();
        buf ^= 1;
    }
    const int lane = tid & 31, half = lane >> 4, nn = lane & 15;
#pragma unroll
    for (int mi = 0; mi < 2; ++mi)
#pragma unroll
        for (int ni = 0; ni < 4; ++ni)
#pragma unroll
            for (int r = 0; r < 8; ++r) {
                int m = m0 + wm + mi * 16 + r + 8 * half;
                int n = n0 + wn + ni * 16 + nn;
                float v = acc[mi][ni][r] + bias[n];
                if (RELU) v = fmaxf(v, 0.0f);
                if (Cf) Cf[(size_t)m * N + n] = v;
                if (Cb) {
                    if (ATTN_LAYOUT) {               // N==D; relayout to [B,H,S,DK]
                        int bb = m >> 10, s = m & (S_ - 1);
                        int hh = n >> 6,  dk = n & (DK_ - 1);
                        Cb[(((size_t)bb * H_ + hh) * S_ + s) * DK_ + dk] = (bf16)v;
                    } else {
                        Cb[(size_t)m * N + n] = (bf16)v;
                    }
                }
            }
}

// ---- flash attention: strict causal (j<i), q==k ----------------------------
__global__ __launch_bounds__(256) void attn_kernel(
    const bf16* __restrict__ Kb, const bf16* __restrict__ Vb,
    bf16* __restrict__ Ob) {
    const int bh = blockIdx.x;
    const int b = bh / H_, h = bh % H_;
    const int wave = threadIdx.x >> 5;
    const int qbase = blockIdx.y * 128 + wave * 16;
    const bf16* Kp = Kb + (size_t)bh * S_ * DK_;
    const bf16* Vp = Vb + (size_t)bh * S_ * DK_;
    __shared__ bf16 pshared[8][16 * 34];
    bf16* pt = &pshared[wave][0];
    const int lane = threadIdx.x & 31, half = lane >> 4, nn = lane & 15;
    const float scale = 0.125f;                       // 1/sqrt(64)
    const float NEG_INF = -__builtin_inff();

    v16bf q0 = load_a_frag(Kp + (size_t)qbase * DK_,      DK_);
    v16bf q1 = load_a_frag(Kp + (size_t)qbase * DK_ + 32, DK_);

    v8f o[4];
#pragma unroll
    for (int i = 0; i < 4; ++i) o[i] = (v8f){};
    float mrow[8], lrowv[8];
#pragma unroll
    for (int r = 0; r < 8; ++r) { mrow[r] = NEG_INF; lrowv[r] = 0.0f; }

    for (int jb = 0; jb < qbase + 16; jb += 32) {
        v8f z = (v8f){};
        v8f c0 = wmma_bf16(q0, load_b_frag(Kp + (size_t)(jb     ) * DK_,      DK_), z);
        c0     = wmma_bf16(q1, load_b_frag(Kp + (size_t)(jb     ) * DK_ + 32, DK_), c0);
        v8f c1 = wmma_bf16(q0, load_b_frag(Kp + (size_t)(jb + 16) * DK_,      DK_), z);
        c1     = wmma_bf16(q1, load_b_frag(Kp + (size_t)(jb + 16) * DK_ + 32, DK_), c1);
#pragma unroll
        for (int r = 0; r < 8; ++r) {
            int i  = qbase + r + 8 * half;
            int j0 = jb + nn, j1 = j0 + 16;
            float s0 = (j0 < i) ? c0[r] * scale : NEG_INF;
            float s1 = (j1 < i) ? c1[r] * scale : NEG_INF;
            float rm = rowmax16(fmaxf(s0, s1));
            float mn = fmaxf(mrow[r], rm);
            bool empty = (mn == NEG_INF);
            float corr = empty ? 1.0f : __expf(mrow[r] - mn);
            float p0 = empty ? 0.0f : __expf(s0 - mn);
            float p1 = empty ? 0.0f : __expf(s1 - mn);
            float rs = rowsum16(p0 + p1);
            lrowv[r] = lrowv[r] * corr + rs;
            mrow[r]  = mn;
#pragma unroll
            for (int nc = 0; nc < 4; ++nc) o[nc][r] *= corr;
            int m = r + 8 * half;
            pt[m * 34 + nn]      = (bf16)p0;
            pt[m * 34 + 16 + nn] = (bf16)p1;
        }
        asm volatile("s_wait_dscnt 0" ::: "memory");
        v16bf pa = load_a_frag(pt, 34);
#pragma unroll
        for (int nc = 0; nc < 4; ++nc)
            o[nc] = wmma_bf16(pa, load_b_frag_km(Vp + (size_t)jb * DK_ + nc * 16, DK_), o[nc]);
    }
#pragma unroll
    for (int nc = 0; nc < 4; ++nc)
#pragma unroll
        for (int r = 0; r < 8; ++r) {
            int i = qbase + r + 8 * half;
            float val = (lrowv[r] > 0.0f) ? o[nc][r] / lrowv[r] : 0.0f;
            Ob[((size_t)b * S_ + i) * D_ + h * DK_ + nc * 16 + nn] = (bf16)val;
        }
}

// ---- fused residual + LayerNorm -------------------------------------------
__global__ __launch_bounds__(256) void ln_kernel(
    const float* __restrict__ x, const float* __restrict__ d,
    const float* __restrict__ s, const float* __restrict__ bb,
    float* __restrict__ xo, bf16* __restrict__ xob) {
    const int row = blockIdx.x, t = threadIdx.x;
    const float* xr = x + (size_t)row * D_;
    const float* dr = d + (size_t)row * D_;
    float v0 = xr[t] + dr[t];
    float v1 = xr[t + 256] + dr[t + 256];
    __shared__ float rs[256], rq[256];
    rs[t] = v0 + v1;
    rq[t] = v0 * v0 + v1 * v1;
    __syncthreads();
#pragma unroll
    for (int off = 128; off > 0; off >>= 1) {
        if (t < off) { rs[t] += rs[t + off]; rq[t] += rq[t + off]; }
        __syncthreads();
    }
    float mean = rs[0] * (1.0f / D_);
    float var  = rq[0] * (1.0f / D_) - mean * mean;
    float inv  = rsqrtf(var + 1e-5f);
    float o0 = (v0 - mean) * inv * s[t]       + bb[t];
    float o1 = (v1 - mean) * inv * s[t + 256] + bb[t + 256];
    xo[(size_t)row * D_ + t]        = o0;
    xo[(size_t)row * D_ + t + 256]  = o1;
    xob[(size_t)row * D_ + t]       = (bf16)o0;
    xob[(size_t)row * D_ + t + 256] = (bf16)o1;
}

// ---- host orchestration ----------------------------------------------------
extern "C" void kernel_launch(void* const* d_in, const int* in_sizes, int n_in,
                              void* d_out, int out_size, void* d_ws, size_t ws_size,
                              hipStream_t stream) {
    const float* q_emb  = (const float*)d_in[0];
    const float* qa_emb = (const float*)d_in[1];
    const float* pe     = (const float*)d_in[2];
    const float* Wk     = (const float*)d_in[3];
    const float* bk     = (const float*)d_in[4];
    const float* Wv     = (const float*)d_in[5];
    const float* bv     = (const float*)d_in[6];
    const float* Wo     = (const float*)d_in[7];
    const float* bo     = (const float*)d_in[8];
    const float* ln1s   = (const float*)d_in[9];
    const float* ln1b   = (const float*)d_in[10];
    const float* W1     = (const float*)d_in[11];
    const float* b1     = (const float*)d_in[12];
    const float* W2     = (const float*)d_in[13];
    const float* b2     = (const float*)d_in[14];
    const float* ln2s   = (const float*)d_in[15];
    const float* ln2b   = (const float*)d_in[16];

    char* ws = (char*)d_ws;
    size_t off = 0;
    auto alloc = [&](size_t bytes) { char* p = ws + off; off += (bytes + 255) & ~(size_t)255; return p; };

    float* xf32   = (float*)alloc((size_t)MS_ * D_   * 4);
    bf16*  x_bf   = (bf16*) alloc((size_t)MS_ * D_   * 2);
    bf16*  y_bf   = (bf16*) alloc((size_t)MS_ * D_   * 2);
    bf16*  k_bf   = (bf16*) alloc((size_t)MS_ * D_   * 2);   // [B,H,S,DK]
    bf16*  v_bf   = (bf16*) alloc((size_t)MS_ * D_   * 2);   // [B,H,S,DK]
    bf16*  att_bf = (bf16*) alloc((size_t)MS_ * D_   * 2);
    float* proj   = (float*)alloc((size_t)MS_ * D_   * 4);
    bf16*  hid_bf = (bf16*) alloc((size_t)MS_ * DFF_ * 2);
    bf16*  Wk_bf  = (bf16*) alloc((size_t)L_ * D_ * D_   * 2);
    bf16*  Wv_bf  = (bf16*) alloc((size_t)L_ * D_ * D_   * 2);
    bf16*  Wo_bf  = (bf16*) alloc((size_t)L_ * D_ * D_   * 2);
    bf16*  W1_bf  = (bf16*) alloc((size_t)L_ * DFF_ * D_ * 2);
    bf16*  W2_bf  = (bf16*) alloc((size_t)L_ * D_ * DFF_ * 2);

    // prep + weight casts
    prep_kernel<<<(size_t)MS_ * D_ / 256, 256, 0, stream>>>(q_emb, qa_emb, pe, xf32, x_bf, y_bf);
    cast_kernel<<<(size_t)L_ * D_ * D_   / 256, 256, 0, stream>>>(Wk, Wk_bf);
    cast_kernel<<<(size_t)L_ * D_ * D_   / 256, 256, 0, stream>>>(Wv, Wv_bf);
    cast_kernel<<<(size_t)L_ * D_ * D_   / 256, 256, 0, stream>>>(Wo, Wo_bf);
    cast_kernel<<<(size_t)L_ * DFF_ * D_ / 256, 256, 0, stream>>>(W1, W1_bf);
    cast_kernel<<<(size_t)L_ * DFF_ * D_ / 256, 256, 0, stream>>>(W2, W2_bf);

    const dim3 gProj(D_ / 128, MS_ / 128);       // (4, 256)
    const dim3 gFfn1(DFF_ / 128, MS_ / 128);     // (16, 256)
    const dim3 gAttn(B_ * H_, S_ / 128);         // (256, 8)

    for (int l = 0; l < L_; ++l) {
        const size_t wD = (size_t)l * D_ * D_;
        const size_t wF = (size_t)l * DFF_ * D_;
        // K (= Q) and V projections, straight into [B,H,S,DK] bf16
        gemm_wmma<false, true><<<gProj, 256, 0, stream>>>(
            x_bf, Wk_bf + wD, bk + l * D_, nullptr, k_bf, MS_, D_, D_);
        gemm_wmma<false, true><<<gProj, 256, 0, stream>>>(
            y_bf, Wv_bf + wD, bv + l * D_, nullptr, v_bf, MS_, D_, D_);
        // causal flash attention (q==k)
        attn_kernel<<<gAttn, 256, 0, stream>>>(k_bf, v_bf, att_bf);
        // output projection
        gemm_wmma<false, false><<<gProj, 256, 0, stream>>>(
            att_bf, Wo_bf + wD, bo + l * D_, proj, nullptr, MS_, D_, D_);
        ln_kernel<<<MS_, 256, 0, stream>>>(xf32, proj, ln1s + l * D_, ln1b + l * D_, xf32, x_bf);
        // FFN
        gemm_wmma<true, false><<<gFfn1, 256, 0, stream>>>(
            x_bf, W1_bf + wF, b1 + l * DFF_, nullptr, hid_bf, MS_, DFF_, D_);
        gemm_wmma<false, false><<<gProj, 256, 0, stream>>>(
            hid_bf, W2_bf + wF, b2 + l * D_, proj, nullptr, MS_, D_, DFF_);
        float* xout = (l == L_ - 1) ? (float*)d_out : xf32;
        ln_kernel<<<MS_, 256, 0, stream>>>(xf32, proj, ln2s + l * D_, ln2b + l * D_, xout, x_bf);
    }
}
